// Block_54133767799245
// MI455X (gfx1250) — matrix operand order repaired and checked by reading, not compiled, not run on hardware
//
#include <hip/hip_runtime.h>
#include <math.h>

// ---------------------------------------------------------------------------
// Types for CDNA5 WMMA (wave32): v_wmma_f32_16x16x32_bf16
// ---------------------------------------------------------------------------
typedef __attribute__((ext_vector_type(16))) __bf16    v16bf;
typedef __attribute__((ext_vector_type(8)))  float     v8f;
typedef __attribute__((ext_vector_type(8)))  unsigned  v8u;

// Tensor Data Mover descriptor groups (CDNA5 ISA)
typedef __attribute__((ext_vector_type(4))) unsigned tdm_u4;
typedef __attribute__((ext_vector_type(8))) int      tdm_i8;
typedef __attribute__((ext_vector_type(4))) int      tdm_i4;

__device__ __forceinline__ v8f wmma_bf16(v16bf a, v16bf b, v8f c) {
  // (neg_a, A, neg_b, B, c_mod, C, reuse_a, reuse_b)
  return __builtin_amdgcn_wmma_f32_16x16x32_bf16(false, a, false, b, (short)0, c,
                                                 false, false);
}

// ---------------------------------------------------------------------------
// TDM: async 2-D tile copy global -> LDS (bf16 elements), one issue per wave.
// D# packing per ISA §8.3/8.4: group0 = {count, lds_addr, global_addr, type=2},
// group1 = {data_size=2B, tensor_dim0=row stride (no OOB), tensor_dim1=rows,
//           tile_dim0=w, tile_dim1=h, dim0_stride=row stride}. Groups 2/3 = 0
// (2-D tensor). Tracked by TENSORcnt.
// ---------------------------------------------------------------------------
__device__ __forceinline__ void tdm_load_2d(const __bf16* lds_dst,
                                            const __bf16* gsrc,
                                            unsigned tile_w_elems,
                                            unsigned tile_h_rows,
                                            unsigned row_stride_elems) {
  const unsigned lds_off =
      (unsigned)(unsigned long long)(uintptr_t)lds_dst;  // addr[31:0] = LDS offset
  const unsigned long long ga = (unsigned long long)(uintptr_t)gsrc;
  tdm_u4 g0;
  g0[0] = 1u;                                                // count=1 (valid D#)
  g0[1] = lds_off;                                           // lds_addr
  g0[2] = (unsigned)ga;                                      // global_addr[31:0]
  g0[3] = (unsigned)((ga >> 32) & 0x01FFFFFFu) | (2u << 30); // [56:32] | type=2
  tdm_i8 g1;
  g1[0] = (int)(1u << 16);                                   // data_size=1 -> 2B
  g1[1] = (int)((row_stride_elems & 0xFFFFu) << 16);         // tensor_dim0 lo
  g1[2] = (int)(((row_stride_elems >> 16) & 0xFFFFu) |       // tensor_dim0 hi
                ((tile_h_rows & 0xFFFFu) << 16));            // tensor_dim1 lo
  g1[3] = (int)(((tile_h_rows >> 16) & 0xFFFFu) |            // tensor_dim1 hi
                ((tile_w_elems & 0xFFFFu) << 16));           // tile_dim0
  g1[4] = (int)(tile_h_rows & 0xFFFFu);                      // tile_dim1 (dim2=0)
  g1[5] = (int)row_stride_elems;                             // dim0_stride[31:0]
  g1[6] = 0;                                                 // stride hi, dim1_stride
  g1[7] = 0;
  const tdm_i4 z4 = {0, 0, 0, 0};
#if defined(__clang_major__) && (__clang_major__ >= 23)
  const tdm_i8 z8 = {0, 0, 0, 0, 0, 0, 0, 0};
  __builtin_amdgcn_tensor_load_to_lds(g0, g1, z4, z4, z8, 0);
#else
  __builtin_amdgcn_tensor_load_to_lds(g0, g1, z4, z4, 0);
#endif
}

__device__ __forceinline__ void tdm_wait_all() {
  __builtin_amdgcn_s_wait_tensorcnt((short)0);
}

// ---------------------------------------------------------------------------
// Fragment loaders (wave32), contiguous k order: element s of lane l holds k = 16*(l/16) + s
// ---------------------------------------------------------------------------

// A-matrix 16x32 (MxK), source row-major [16][ldk] (bf16).
// lanes 0-15: M=lane, VGPR v holds K = {2v,2v+1}
// lanes 16-31: same +16.
__device__ __forceinline__ v16bf load_a_frag(const __bf16* base, int ldk,
                                             int kOff, int lane) {
  const int m     = lane & 15;
  const int khalf = (lane & 16) ? 16 : 0;
  const unsigned* rw = reinterpret_cast<const unsigned*>(base + (size_t)m * ldk);
  v8u w;
#pragma unroll
  for (int v = 0; v < 8; ++v) {
    const int k = kOff + 2 * v + khalf;
    w[v] = rw[k >> 1];
  }
  return __builtin_bit_cast(v16bf, w);
}

// B-matrix 32x16 (KxN) from row-major [K][ldn] (bf16); strided scalar loads.
// lane%16 = N, element e -> K = kOff + e (+16 for lanes 16-31).
__device__ __forceinline__ v16bf load_b_frag(const __bf16* base, int ldn,
                                             int n0, int kOff, int lane) {
  const int n  = n0 + (lane & 15);
  const int kb = kOff + ((lane & 16) ? 16 : 0);
  v16bf b = {};
#pragma unroll
  for (int e = 0; e < 16; ++e) b[e] = base[(size_t)(kb + e) * ldn + n];
  return b;
}

// B-matrix 32x16 from K-contiguous (transposed) storage: row-major [N][ldk];
// consecutive e == consecutive K -> contiguous dword loads (2x ds_load_b128).
__device__ __forceinline__ v16bf load_bt_frag(const __bf16* base, int ldk,
                                              int n0, int kOff, int lane) {
  const int n  = n0 + (lane & 15);
  const int k0 = kOff + ((lane & 16) ? 16 : 0);
  const unsigned* rw =
      reinterpret_cast<const unsigned*>(base + (size_t)n * ldk + k0);
  v8u w;
#pragma unroll
  for (int e = 0; e < 8; ++e) w[e] = rw[e];
  return __builtin_bit_cast(v16bf, w);
}

__device__ __forceinline__ float gelu_f(float v) {
  const float c = 0.7978845608028654f;  // sqrt(2/pi)
  const float t = tanhf(c * (v + 0.044715f * v * v * v));
  return 0.5f * v * (1.0f + t);
}

// ---------------------------------------------------------------------------
// fp32 [K][N] -> bf16 transposed [N][K] (weights). 32x32 LDS tile transpose;
// makes the GEMM B operand K-contiguous so B-frags load as 2x ds_load_b128.
// ---------------------------------------------------------------------------
__global__ void __launch_bounds__(256)
cvt_transpose_bf16_kernel(const float* __restrict__ in, __bf16* __restrict__ out,
                          int K, int N) {
  __shared__ __bf16 tile[32][33];
  const int n0 = blockIdx.x * 32;
  const int k0 = blockIdx.y * 32;
  const int c  = threadIdx.x & 31;
  const int r4 = (threadIdx.x >> 5) * 4;
#pragma unroll
  for (int i = 0; i < 4; ++i) {
    const int r = r4 + i;  // K direction
    tile[r][c] = (__bf16)in[(size_t)(k0 + r) * N + n0 + c];
  }
  __syncthreads();
#pragma unroll
  for (int i = 0; i < 4; ++i) {
    const int r = r4 + i;  // N direction on write
    out[(size_t)(n0 + r) * K + k0 + c] = tile[c][r];
  }
}

// ---------------------------------------------------------------------------
// LayerNorm (one row per block), fp32 in -> bf16 out
// ---------------------------------------------------------------------------
__global__ void __launch_bounds__(256)
layernorm_bf16_kernel(const float* __restrict__ x, const float* __restrict__ g,
                      const float* __restrict__ bta, __bf16* __restrict__ out,
                      int C_) {
  const float* xr = x + (size_t)blockIdx.x * C_;
  float s = 0.f, ss = 0.f;
  for (int i = threadIdx.x; i < C_; i += 256) {
    const float v = xr[i];
    s += v;
    ss += v * v;
  }
#pragma unroll
  for (int o = 16; o > 0; o >>= 1) {
    s  += __shfl_xor(s, o, 32);
    ss += __shfl_xor(ss, o, 32);
  }
  __shared__ float ps[8], pss[8];
  if ((threadIdx.x & 31) == 0) {
    ps[threadIdx.x >> 5]  = s;
    pss[threadIdx.x >> 5] = ss;
  }
  __syncthreads();
  float ts = 0.f, tss = 0.f;
#pragma unroll
  for (int w = 0; w < 8; ++w) { ts += ps[w]; tss += pss[w]; }
  const float mu   = ts / C_;
  const float var  = tss / C_ - mu * mu;
  const float rstd = rsqrtf(var + 1e-5f);
  for (int i = threadIdx.x; i < C_; i += 256) {
    const float v = (xr[i] - mu) * rstd * g[i] + bta[i];
    out[(size_t)blockIdx.x * C_ + i] = (__bf16)v;
  }
}

// ---------------------------------------------------------------------------
// Tiled bf16 GEMM, fp32 accumulate: D = epi(A[M,K] * Wt^T + bias (+resid)),
// with Wt given TRANSPOSED: Wt[N][K]. 256 threads = 8 waves; tile 128x128x32;
// wave grid 2x4, per-wave 4x2 frags. TDM double-buffered staging: wave 0
// issues tensor_load_to_lds for tile i+1 while all waves run WMMAs on tile i.
// Both LDS tiles are 128 rows x 32 K (row stride = K in global).
// EPI: 0 = bias -> bf16, 1 = bias+GELU -> bf16, 2 = bias+resid -> f32
// ---------------------------------------------------------------------------
template <int EPI>
__global__ void __launch_bounds__(256)
gemm_bf16_kernel(const __bf16* __restrict__ A, const __bf16* __restrict__ Wt,
                 const float* __restrict__ bias, const float* __restrict__ resid,
                 __bf16* __restrict__ outB, float* __restrict__ outF,
                 int M, int N, int K) {
  __shared__ __attribute__((aligned(16))) __bf16 sA[2][128 * 32];
  __shared__ __attribute__((aligned(16))) __bf16 sB[2][128 * 32];

  const int tid   = threadIdx.x;
  const int lane  = tid & 31;
  const int wid   = tid >> 5;
  const int wm    = wid >> 2;  // 0..1  (64 rows each)
  const int wn    = wid & 3;   // 0..3  (32 cols each)
  const int mBase = blockIdx.y * 128;
  const int nBase = blockIdx.x * 128;

  const v8f vzero = {0.f, 0.f, 0.f, 0.f, 0.f, 0.f, 0.f, 0.f};
  v8f acc[4][2];
#pragma unroll
  for (int i = 0; i < 4; ++i)
#pragma unroll
    for (int j = 0; j < 2; ++j) acc[i][j] = vzero;

  const int nIter = K / 32;

  // Prologue: DMA first tiles into buffer 0
  if (wid == 0) {
    tdm_load_2d(&sA[0][0], A + (size_t)mBase * K, 32, 128, (unsigned)K);
    tdm_load_2d(&sB[0][0], Wt + (size_t)nBase * K, 32, 128, (unsigned)K);
  }

  for (int i = 0; i < nIter; ++i) {
    const int cur = i & 1;
    if (wid == 0) tdm_wait_all();  // buf[cur] DMA complete (in-order TDM)
    __syncthreads();               // publish buf[cur] to all waves
    if (wid == 0 && i + 1 < nIter) {
      const int k1 = (i + 1) * 32;  // overlap DMA of tile i+1 with compute
      tdm_load_2d(&sA[cur ^ 1][0], A + (size_t)mBase * K + k1, 32, 128,
                  (unsigned)K);
      tdm_load_2d(&sB[cur ^ 1][0], Wt + (size_t)nBase * K + k1, 32, 128,
                  (unsigned)K);
    }

    v16bf af[4], bf[2];
#pragma unroll
    for (int mi = 0; mi < 4; ++mi)
      af[mi] = load_a_frag(&sA[cur][(wm * 64 + mi * 16) * 32], 32, 0, lane);
#pragma unroll
    for (int ni = 0; ni < 2; ++ni)
      bf[ni] = load_bt_frag(&sB[cur][0], 32, wn * 32 + ni * 16, 0, lane);

#pragma unroll
    for (int mi = 0; mi < 4; ++mi)
#pragma unroll
      for (int ni = 0; ni < 2; ++ni)
        acc[mi][ni] = wmma_bf16(af[mi], bf[ni], acc[mi][ni]);

    __syncthreads();               // retire reads of buf[cur] before refill
  }

  // Epilogue: C-frag layout -> lane%16 = col, VGPR r -> row r + 8*(lane/16)
#pragma unroll
  for (int mi = 0; mi < 4; ++mi) {
#pragma unroll
    for (int ni = 0; ni < 2; ++ni) {
      const int col = nBase + wn * 32 + ni * 16 + (lane & 15);
      const float bv = bias[col];
#pragma unroll
      for (int r = 0; r < 8; ++r) {
        const int row = mBase + wm * 64 + mi * 16 + r + ((lane & 16) ? 8 : 0);
        float v = acc[mi][ni][r] + bv;
        const size_t idx = (size_t)row * N + col;
        if constexpr (EPI == 0) {
          outB[idx] = (__bf16)v;
        } else if constexpr (EPI == 1) {
          outB[idx] = (__bf16)gelu_f(v);
        } else {
          outF[idx] = v + resid[idx];
        }
      }
    }
  }
}

// ---------------------------------------------------------------------------
// Flash attention (non-causal). 256 threads = 8 waves; each wave owns a
// 16-row query strip; block covers 128 query rows of one (b,h). K/V staged in
// 64-key LDS tiles via double-buffered TDM. QK^T and P*V use WMMA bf16.
// ---------------------------------------------------------------------------
__global__ void __launch_bounds__(256)
attention_kernel(const __bf16* __restrict__ qkv, __bf16* __restrict__ y,
                 int Bc, int Tc, int Cc, int Hc) {
  const int hd      = 64;
  const int qblocks = Tc / 128;
  const int bid = blockIdx.x;
  const int qb  = bid % qblocks;
  const int h   = (bid / qblocks) % Hc;
  const int b   = bid / (qblocks * Hc);

  const int tid  = threadIdx.x;
  const int lane = tid & 31;
  const int wid  = tid >> 5;
  const int qrow0 = qb * 128 + wid * 16;
  const int C3 = 3 * Cc;

  __shared__ __attribute__((aligned(16))) __bf16 sK[2][64 * 64];
  __shared__ __attribute__((aligned(16))) __bf16 sV[2][64 * 64];
  __shared__ __attribute__((aligned(16))) __bf16 sP[8][16 * 64];
  __bf16* sPw = &sP[wid][0];

  // Q fragments for this wave's 16 rows (hd=64 -> two 16x32 A-frags)
  const __bf16* qbase = qkv + (size_t)(b * Tc + qrow0) * C3 + h * hd;
  v16bf qf[2];
  qf[0] = load_a_frag(qbase, C3, 0, lane);
  qf[1] = load_a_frag(qbase, C3, 32, lane);

  const v8f vzero = {0.f, 0.f, 0.f, 0.f, 0.f, 0.f, 0.f, 0.f};
  v8f o[4];
#pragma unroll
  for (int nf = 0; nf < 4; ++nf) o[nf] = vzero;
  float mrow[8], lrow[8];
#pragma unroll
  for (int r = 0; r < 8; ++r) { mrow[r] = -1e30f; lrow[r] = 0.f; }

  const float scale = 0.125f;  // 1/sqrt(64)
  const int nkb = Tc / 64;

  // Prologue: DMA K/V tiles for key-block 0
  if (wid == 0) {
    const __bf16* kv0 = qkv + (size_t)(b * Tc) * C3 + h * hd;
    tdm_load_2d(&sK[0][0], kv0 + Cc, 64, 64, (unsigned)C3);
    tdm_load_2d(&sV[0][0], kv0 + 2 * Cc, 64, 64, (unsigned)C3);
  }

  for (int kb = 0; kb < nkb; ++kb) {
    const int cur = kb & 1;
    if (wid == 0) tdm_wait_all();
    __syncthreads();  // publish K/V tile kb
    if (wid == 0 && kb + 1 < nkb) {
      const __bf16* kvn =
          qkv + (size_t)(b * Tc + (kb + 1) * 64) * C3 + h * hd;
      tdm_load_2d(&sK[cur ^ 1][0], kvn + Cc, 64, 64, (unsigned)C3);
      tdm_load_2d(&sV[cur ^ 1][0], kvn + 2 * Cc, 64, 64, (unsigned)C3);
    }

    // S = Q * K^T : 16 x 64 per wave. Preload all 4 B-frags per k-step so the
    // 4 WMMAs issue back-to-back (no per-WMMA DS wait).
    v8f sacc[4];
#pragma unroll
    for (int nf = 0; nf < 4; ++nf) sacc[nf] = vzero;
#pragma unroll
    for (int kk = 0; kk < 2; ++kk) {
      v16bf kf[4];
#pragma unroll
      for (int nf = 0; nf < 4; ++nf)
        kf[nf] = load_bt_frag(&sK[cur][0], 64, nf * 16, kk * 32, lane);
#pragma unroll
      for (int nf = 0; nf < 4; ++nf)
        sacc[nf] = wmma_bf16(qf[kk], kf[nf], sacc[nf]);
    }

    // Online softmax. C-frag row r + 8*(lane/16) lives in a 16-lane half-group.
    float mnew[8], corr[8];
#pragma unroll
    for (int r = 0; r < 8; ++r) {
      float mx = -1e30f;
#pragma unroll
      for (int nf = 0; nf < 4; ++nf) mx = fmaxf(mx, sacc[nf][r]);
      mx *= scale;
#pragma unroll
      for (int off = 8; off > 0; off >>= 1)
        mx = fmaxf(mx, __shfl_xor(mx, off, 32));
      mnew[r] = fmaxf(mrow[r], mx);
      corr[r] = __expf(mrow[r] - mnew[r]);
      mrow[r] = mnew[r];
    }

    float rsum[8];
#pragma unroll
    for (int r = 0; r < 8; ++r) rsum[r] = 0.f;
#pragma unroll
    for (int nf = 0; nf < 4; ++nf) {
      const int col = nf * 16 + (lane & 15);
#pragma unroll
      for (int r = 0; r < 8; ++r) {
        const float p = __expf(sacc[nf][r] * scale - mnew[r]);
        rsum[r] += p;
        sPw[(r + ((lane & 16) ? 8 : 0)) * 64 + col] = (__bf16)p;
      }
    }
#pragma unroll
    for (int r = 0; r < 8; ++r) {
      float s = rsum[r];
#pragma unroll
      for (int off = 8; off > 0; off >>= 1) s += __shfl_xor(s, off, 32);
      lrow[r] = lrow[r] * corr[r] + s;
    }
#pragma unroll
    for (int nf = 0; nf < 4; ++nf)
#pragma unroll
      for (int r = 0; r < 8; ++r) o[nf][r] *= corr[r];

    // O += P * V : contraction over 64 keys in 2 steps of 32; preload V-frags.
    // (per-wave P strip: same-wave LDS ops are in-order, no barrier needed)
#pragma unroll
    for (int kk = 0; kk < 2; ++kk) {
      v16bf pa = load_a_frag(sPw, 64, kk * 32, lane);
      v16bf vb[4];
#pragma unroll
      for (int nf = 0; nf < 4; ++nf)
        vb[nf] = load_b_frag(&sV[cur][0], 64, nf * 16, kk * 32, lane);
#pragma unroll
      for (int nf = 0; nf < 4; ++nf) o[nf] = wmma_bf16(pa, vb[nf], o[nf]);
    }

    __syncthreads();  // retire reads of tile kb before refill
  }

  // Normalize and write y (bf16) at head column offset
#pragma unroll
  for (int nf = 0; nf < 4; ++nf) {
    const int col = h * hd + nf * 16 + (lane & 15);
#pragma unroll
    for (int r = 0; r < 8; ++r) {
      const int row = qrow0 + r + ((lane & 16) ? 8 : 0);
      const float val = o[nf][r] / lrow[r];
      y[(size_t)(b * Tc + row) * Cc + col] = (__bf16)val;
    }
  }
}

// ---------------------------------------------------------------------------
// Host-side orchestration
// ---------------------------------------------------------------------------
extern "C" void kernel_launch(void* const* d_in, const int* in_sizes, int n_in,
                              void* d_out, int out_size, void* d_ws,
                              size_t ws_size, hipStream_t stream) {
  (void)in_sizes; (void)n_in; (void)out_size; (void)ws_size;

  const float* x      = (const float*)d_in[0];
  const float* ln1_g  = (const float*)d_in[1];
  const float* ln1_b  = (const float*)d_in[2];
  const float* w_qkv  = (const float*)d_in[3];
  const float* b_qkv  = (const float*)d_in[4];
  const float* w_ap   = (const float*)d_in[5];
  const float* b_ap   = (const float*)d_in[6];
  const float* ln2_g  = (const float*)d_in[7];
  const float* ln2_b  = (const float*)d_in[8];
  const float* w_fc   = (const float*)d_in[9];
  const float* b_fc   = (const float*)d_in[10];
  const float* w_proj = (const float*)d_in[11];
  const float* b_proj = (const float*)d_in[12];
  float* out = (float*)d_out;

  constexpr int Bc = 4, Tc = 2048, Cc = 1024, Hc = 16, F = 4096;
  constexpr int BT = Bc * Tc;

  char* ws = (char*)d_ws;
  size_t off = 0;
  auto alloc = [&](size_t elems, size_t esz) -> void* {
    void* p = ws + off;
    off += (elems * esz + 255) & ~(size_t)255;
    return p;
  };
  // Weights stored TRANSPOSED ([N][K]) in bf16 for K-contiguous B-frag loads.
  __bf16* wqkv_t  = (__bf16*)alloc((size_t)Cc * 3 * Cc, 2);
  __bf16* wap_t   = (__bf16*)alloc((size_t)Cc * Cc, 2);
  __bf16* wfc_t   = (__bf16*)alloc((size_t)Cc * F, 2);
  __bf16* wproj_t = (__bf16*)alloc((size_t)F * Cc, 2);
  __bf16* h_b     = (__bf16*)alloc((size_t)BT * Cc, 2);
  __bf16* qkv_b   = (__bf16*)alloc((size_t)BT * 3 * Cc, 2);
  __bf16* y_b     = (__bf16*)alloc((size_t)BT * Cc, 2);
  float*  x2      = (float*)alloc((size_t)BT * Cc, 4);
  __bf16* h2_b    = (__bf16*)alloc((size_t)BT * Cc, 2);
  __bf16* hg_b    = (__bf16*)alloc((size_t)BT * F, 2);

  const dim3 blk(256);

  // Weight conversions fp32 [K][N] -> bf16 transposed [N][K]
  cvt_transpose_bf16_kernel<<<dim3(3 * Cc / 32, Cc / 32), blk, 0, stream>>>(
      w_qkv, wqkv_t, Cc, 3 * Cc);
  cvt_transpose_bf16_kernel<<<dim3(Cc / 32, Cc / 32), blk, 0, stream>>>(
      w_ap, wap_t, Cc, Cc);
  cvt_transpose_bf16_kernel<<<dim3(F / 32, Cc / 32), blk, 0, stream>>>(
      w_fc, wfc_t, Cc, F);
  cvt_transpose_bf16_kernel<<<dim3(Cc / 32, F / 32), blk, 0, stream>>>(
      w_proj, wproj_t, F, Cc);

  // h = LN1(x)
  layernorm_bf16_kernel<<<dim3(BT), blk, 0, stream>>>(x, ln1_g, ln1_b, h_b, Cc);

  // qkv = h @ w_qkv + b_qkv          [BT, 3C] bf16
  gemm_bf16_kernel<0><<<dim3(3 * Cc / 128, BT / 128), blk, 0, stream>>>(
      h_b, wqkv_t, b_qkv, nullptr, qkv_b, nullptr, BT, 3 * Cc, Cc);

  // y = attention(q, k, v)           [BT, C] bf16
  attention_kernel<<<dim3(Bc * Hc * (Tc / 128)), blk, 0, stream>>>(
      qkv_b, y_b, Bc, Tc, Cc, Hc);

  // x2 = x + y @ w_attnproj + b      [BT, C] f32
  gemm_bf16_kernel<2><<<dim3(Cc / 128, BT / 128), blk, 0, stream>>>(
      y_b, wap_t, b_ap, x, nullptr, x2, BT, Cc, Cc);

  // h2 = LN2(x2)
  layernorm_bf16_kernel<<<dim3(BT), blk, 0, stream>>>(x2, ln2_g, ln2_b, h2_b, Cc);

  // hg = gelu(h2 @ w_fc + b_fc)      [BT, 4C] bf16
  gemm_bf16_kernel<1><<<dim3(F / 128, BT / 128), blk, 0, stream>>>(
      h2_b, wfc_t, b_fc, nullptr, hg_b, nullptr, BT, F, Cc);

  // out = x2 + hg @ w_proj + b_proj  [BT, C] f32
  gemm_bf16_kernel<2><<<dim3(Cc / 128, BT / 128), blk, 0, stream>>>(
      hg_b, wproj_t, b_proj, x2, nullptr, out, BT, Cc, F);
}
